// CodeBook_3539053052737
// MI455X (gfx1250) — compile-verified, matrix operand order checked by reading, not run on hardware
//
#include <hip/hip_runtime.h>
#include <hip/hip_bf16.h>

// ---------------------------------------------------------------------------
// VQ-VAE codebook quantization for MI455X (gfx1250, wave32, WMMA + async LDS)
//   z:        [32, 256, 32, 32] f32   (B, D, H, W)
//   codebook: [1024, 256] f32         (K, D)
// outputs (concatenated in d_out):
//   z_q  [32,256,32,32] (8388608 f32), idx [32768] (int32 bits), loss [1 f32]
// ---------------------------------------------------------------------------

typedef __attribute__((ext_vector_type(16))) __bf16 v16bf;
typedef __attribute__((ext_vector_type(8)))  float  v8f;

#define K_CODES        1024
#define D_DIM          256
#define N_ROWS         32768            // 32 * 32 * 32
#define ZQ_ELEMS       8388608          // 32 * 256 * 32 * 32
#define ROWS_PER_BLOCK 128
#define CHUNK_CODES    64               // codes per LDS chunk (ping-pong x2)
#define NUM_CHUNKS     (K_CODES / CHUNK_CODES)
#define LDS_STRIDE     272              // 256 + 16 bf16 pad (breaks bank conflicts)

// 1 = use GLOBAL_LOAD_ASYNC_TO_LDS_B128 (ASYNCcnt) for codebook staging
#ifndef USE_ASYNC
#define USE_ASYNC 1
#endif

// workspace layout (bytes)
#define WS_CB_BF16     0                // 1024*256*2 = 524288
#define WS_ENORM       524288           // 1024*4     = 4096
#define WS_IDX         528384           // 32768*4    = 131072
#define WS_ACC         659456           // 8 (double)

// ---------------------------------------------------------------------------
// Kernel 0: codebook f32 -> bf16, per-code ||e||^2, zero loss accumulator.
// ---------------------------------------------------------------------------
__global__ __launch_bounds__(256)
void prep_codebook(const float* __restrict__ cb,
                   __bf16* __restrict__ cb_bf,
                   float* __restrict__ enorm,
                   double* __restrict__ acc)
{
    __shared__ float red[8];
    const int c = blockIdx.x;
    const int t = threadIdx.x;
    const float v = cb[c * D_DIM + t];
    cb_bf[c * D_DIM + t] = (__bf16)v;
    float s = v * v;
    #pragma unroll
    for (int m = 16; m >= 1; m >>= 1) s += __shfl_xor(s, m, 32);
    if ((t & 31) == 0) red[t >> 5] = s;
    __syncthreads();
    if (t == 0) {
        float tot = 0.f;
        #pragma unroll
        for (int i = 0; i < 8; ++i) tot += red[i];
        enorm[c] = tot;
        if (c == 0) *acc = 0.0;
    }
}

// ---------------------------------------------------------------------------
// Async (memory -> LDS, no VGPR data) or sync staging of one codebook chunk.
// 64 codes x 256 bf16 = 2048 x 16B units; 256 threads -> 8 units each.
// ---------------------------------------------------------------------------
__device__ __forceinline__ void stage_chunk(const __bf16* __restrict__ cb_bf,
                                            __bf16* dstbuf, int chunk, int t)
{
    const uint4* src = (const uint4*)(cb_bf + chunk * CHUNK_CODES * D_DIM);
    #pragma unroll
    for (int i = 0; i < (CHUNK_CODES * 32) / 256; ++i) {    // 8 units / thread
        const int u    = i * 256 + t;
        const int code = u >> 5;
        const int k16  = u & 31;
        __bf16* dst = &dstbuf[code * LDS_STRIDE + k16 * 8];
#if USE_ASYNC
        // GLOBAL_LOAD_ASYNC_TO_LDS_B128, GV mode: VDST = wave-relative LDS
        // byte address (low 32 bits of the generic LDS pointer), VADDR = 64b
        // global address. Tracked by ASYNCcnt.
        const unsigned int       ldsoff = (unsigned int)(unsigned long long)(uintptr_t)dst;
        const unsigned long long gaddr  = (unsigned long long)(uintptr_t)(src + u);
        asm volatile("global_load_async_to_lds_b128 %0, %1, off"
                     :: "v"(ldsoff), "v"(gaddr) : "memory");
#else
        *(uint4*)dst = src[u];
#endif
    }
}

__device__ __forceinline__ void wait_async_zero()
{
#if USE_ASYNC
    asm volatile("s_wait_asynccnt 0x0" ::: "memory");
#endif
}

// ---------------------------------------------------------------------------
// Kernel 1: fused distance GEMM (bf16 WMMA) + per-row argmin.
// score = ||e||^2 - 2 * (z . e)   (||z||^2 is row-constant -> drops out)
// grid = 256 blocks (128 z-rows each), block = 256 threads (8 waves).
// LDS ping-pong on the codebook chunk: async loads of chunk c+1 overlap the
// 32 WMMAs of chunk c. B fragments double-buffered in VGPRs inside the
// K-loop so ds_load latency hides under the previous WMMA.
// ---------------------------------------------------------------------------
__global__ __launch_bounds__(256)
void vq_argmin(const float* __restrict__ z,
               const __bf16* __restrict__ cb_bf,
               const float* __restrict__ enorm,
               int* __restrict__ idx_ws,
               int* __restrict__ idx_out)
{
    __shared__ __bf16 lds_a[ROWS_PER_BLOCK * LDS_STRIDE];     // 69632 B
    __shared__ __bf16 lds_b[2][CHUNK_CODES * LDS_STRIDE];     // 2x34816 B
    __shared__ float  lds_en[K_CODES];                        // 4096 B

    const int t        = threadIdx.x;
    const int row_base = blockIdx.x * ROWS_PER_BLOCK;
    const int b        = row_base >> 10;        // batch index (1024 rows/batch)
    const int hw_base  = row_base & 1023;

    // stage all ||e||^2
    for (int i = t; i < K_CODES; i += 256) lds_en[i] = enorm[i];

    // kick off async staging of chunk 0 while we convert the A tile
    stage_chunk(cb_bf, &lds_b[0][0], 0, t);

    // stage A tile: 128 rows (hw) x 256 (d), f32 -> bf16, float4 along hw.
    for (int u = t; u < (ROWS_PER_BLOCK / 4) * D_DIM; u += 256) {
        const int d   = u >> 5;                 // 32 float4 units per d
        const int hw4 = (u & 31) << 2;
        const float4 v = *(const float4*)&z[((size_t)(b * D_DIM + d) << 10) + hw_base + hw4];
        lds_a[(hw4 + 0) * LDS_STRIDE + d] = (__bf16)v.x;
        lds_a[(hw4 + 1) * LDS_STRIDE + d] = (__bf16)v.y;
        lds_a[(hw4 + 2) * LDS_STRIDE + d] = (__bf16)v.z;
        lds_a[(hw4 + 3) * LDS_STRIDE + d] = (__bf16)v.w;
    }
    __syncthreads();

    const int wave = t >> 5;
    const int lane = t & 31;
    const int col  = lane & 15;      // N column within 16x16 tile
    const int half = lane >> 4;      // 0: K lo half, 1: K hi half
    const int arow = wave * 16 + col;

    union Frag { v16bf v; uint4 q[2]; };

    // preload 8 A fragments (full K=256): 16 bf16 per lane each
    Frag a[8];
    #pragma unroll
    for (int kf = 0; kf < 8; ++kf) {
        const uint4* p = (const uint4*)&lds_a[arow * LDS_STRIDE + kf * 32 + half * 16];
        a[kf].q[0] = p[0];
        a[kf].q[1] = p[1];
    }

    float best[8];
    int   bidx[8];
    #pragma unroll
    for (int r = 0; r < 8; ++r) { best[r] = 3.4e38f; bidx[r] = 0; }

    for (int chunk = 0; chunk < NUM_CHUNKS; ++chunk) {
        wait_async_zero();               // this wave's async loads landed
        __syncthreads();                 // everyone's landed; prev compute done
        if (chunk + 1 < NUM_CHUNKS)
            stage_chunk(cb_bf, &lds_b[(chunk + 1) & 1][0], chunk + 1, t);

        const __bf16* bufc = &lds_b[chunk & 1][0];
        #pragma unroll
        for (int nt = 0; nt < CHUNK_CODES / 16; ++nt) {
            const __bf16* bbase = bufc + (nt * 16 + col) * LDS_STRIDE + half * 16;
            Frag bb[2];
            bb[0].q[0] = ((const uint4*)bbase)[0];
            bb[0].q[1] = ((const uint4*)bbase)[1];
            v8f acc = {0.f, 0.f, 0.f, 0.f, 0.f, 0.f, 0.f, 0.f};
            #pragma unroll
            for (int kf = 0; kf < 8; ++kf) {
                if (kf < 7) {            // prefetch next B fragment
                    const uint4* pn = (const uint4*)(bbase + (kf + 1) * 32);
                    bb[(kf + 1) & 1].q[0] = pn[0];
                    bb[(kf + 1) & 1].q[1] = pn[1];
                }
                acc = __builtin_amdgcn_wmma_f32_16x16x32_bf16(
                    /*neg_a=*/false, a[kf].v, /*neg_b=*/false, bb[kf & 1].v,
                    /*c_mod=*/(short)0, acc, /*reuse_a=*/false, /*reuse_b=*/false);
            }
            const int   code = chunk * CHUNK_CODES + nt * 16 + col;
            const float en   = lds_en[code];
            // C layout: VGPR r -> row (r + 8*half), lane&15 -> column
            #pragma unroll
            for (int r = 0; r < 8; ++r) {
                const float s = en - 2.0f * acc[r];
                if (s < best[r]) { best[r] = s; bidx[r] = code; }
            }
        }
    }

    // per-row argmin: min-reduce across the 16 lanes of each half-wave
    #pragma unroll
    for (int r = 0; r < 8; ++r) {
        float v  = best[r];
        int   ix = bidx[r];
        #pragma unroll
        for (int m = 8; m >= 1; m >>= 1) {
            const float ov = __shfl_xor(v, m, 16);
            const int   oi = __shfl_xor(ix, m, 16);
            if (ov < v || (ov == v && oi < ix)) { v = ov; ix = oi; }
        }
        if (col == 0) {
            const int row = row_base + wave * 16 + half * 8 + r;
            idx_ws[row]  = ix;
            idx_out[row] = ix;
        }
    }
}

// ---------------------------------------------------------------------------
// Kernel 2: z_q write in [B,D,H,W] order + loss accumulation.
// straight-through: z_q == codebook[idx] numerically. Coalesced stores; the
// codebook gather is L2-resident (1 MB << 192 MB L2).
// ---------------------------------------------------------------------------
__global__ __launch_bounds__(256)
void vq_output(const float* __restrict__ z,
               const float* __restrict__ cb,
               const int* __restrict__ idx,
               float* __restrict__ zq,
               double* __restrict__ acc)
{
    __shared__ float red[8];
    const int t = threadIdx.x;
    const size_t base = (size_t)blockIdx.x * 2048;
    float lsum = 0.f;
    #pragma unroll
    for (int i = 0; i < 8; ++i) {
        const size_t l = base + (size_t)i * 256 + t;
        const int d  = (int)((l >> 10) & 255);
        const int hw = (int)(l & 1023);
        const int bb = (int)(l >> 18);
        const int n  = (bb << 10) | hw;
        const float v = cb[idx[n] * D_DIM + d];
        const float diff = v - z[l];
        zq[l] = v;
        lsum += diff * diff;
    }
    #pragma unroll
    for (int m = 16; m >= 1; m >>= 1) lsum += __shfl_xor(lsum, m, 32);
    if ((t & 31) == 0) red[t >> 5] = lsum;
    __syncthreads();
    if (t == 0) {
        float tot = 0.f;
        #pragma unroll
        for (int i = 0; i < 8; ++i) tot += red[i];
        atomicAdd(acc, (double)tot);
    }
}

// loss = mean + BETA*mean = 1.25 * sum / (N*D)
__global__ void vq_finalize(const double* __restrict__ acc, float* __restrict__ loss)
{
    *loss = (float)(1.25 * (*acc) / (double)ZQ_ELEMS);
}

// ---------------------------------------------------------------------------
extern "C" void kernel_launch(void* const* d_in, const int* in_sizes, int n_in,
                              void* d_out, int out_size, void* d_ws, size_t ws_size,
                              hipStream_t stream)
{
    const float* z  = (const float*)d_in[0];   // [32,256,32,32]
    const float* cb = (const float*)d_in[1];   // [1024,256]
    float* out = (float*)d_out;

    char* ws = (char*)d_ws;                    // needs ~660 KB
    __bf16* cb_bf  = (__bf16*)(ws + WS_CB_BF16);
    float*  enorm  = (float*)(ws + WS_ENORM);
    int*    idxbuf = (int*)(ws + WS_IDX);
    double* acc    = (double*)(ws + WS_ACC);

    prep_codebook<<<K_CODES, 256, 0, stream>>>(cb, cb_bf, enorm, acc);
    vq_argmin<<<N_ROWS / ROWS_PER_BLOCK, 256, 0, stream>>>(
        z, cb_bf, enorm, idxbuf, (int*)(out + ZQ_ELEMS));
    vq_output<<<ZQ_ELEMS / 2048, 256, 0, stream>>>(z, cb, idxbuf, out, acc);
    vq_finalize<<<1, 1, 0, stream>>>(acc, out + ZQ_ELEMS + N_ROWS);
}
